// Attention_19524921327902
// MI455X (gfx1250) — matrix operand order repaired and checked by reading, not compiled, not run on hardware
//
#include <hip/hip_runtime.h>
#include <hip/hip_bf16.h>
#include <cstdint>

#define B_   2
#define C_   768
#define S_   2048
#define NH_  8
#define HD_  96
#define NC3_ 2304

typedef __attribute__((ext_vector_type(16))) __bf16 v16bf;
typedef __attribute__((ext_vector_type(8)))  __bf16 v8bf;
typedef __attribute__((ext_vector_type(8)))  float  v8f;
typedef __attribute__((ext_vector_type(4))) unsigned int u32x4;
typedef __attribute__((ext_vector_type(8))) int         i32x8;
typedef __attribute__((ext_vector_type(4))) int         i32x4;

// Row reductions across the 16 lanes holding one C/D row (xor masks 1,2,4,8
// stay inside a 16-lane half, matching the 16x16 f32 D layout).
__device__ __forceinline__ float rowmax16(float v) {
    v = fmaxf(v, __shfl_xor(v, 1, 32));
    v = fmaxf(v, __shfl_xor(v, 2, 32));
    v = fmaxf(v, __shfl_xor(v, 4, 32));
    v = fmaxf(v, __shfl_xor(v, 8, 32));
    return v;
}
__device__ __forceinline__ float rowsum16(float v) {
    v += __shfl_xor(v, 1, 32);
    v += __shfl_xor(v, 2, 32);
    v += __shfl_xor(v, 4, 32);
    v += __shfl_xor(v, 8, 32);
    return v;
}

__device__ __forceinline__ v8f wmma_acc(v16bf a, v16bf b, v8f c) {
    return __builtin_amdgcn_wmma_f32_16x16x32_bf16(false, a, false, b,
                                                   (short)0, c, false, false);
}

// ---------------------------------------------------------------------------
// Kernel 0: weight convert + transpose, f32 (K,N) -> bf16 (N,K).
// Amortizes the f32->bf16 conversion once instead of per s-tile, and makes
// every WMMA B-fragment a single contiguous 32-byte load.
// ---------------------------------------------------------------------------
__global__ __launch_bounds__(256) void wcvt_kernel(
    const float* __restrict__ W, __bf16* __restrict__ Wt, int K, int N)
{
    const int idx = blockIdx.x * 256 + threadIdx.x;
    if (idx < K * N) {
        const int k = idx / N;
        const int n = idx % N;
        Wt[(size_t)n * K + k] = (__bf16)W[idx];
    }
}

// ---------------------------------------------------------------------------
// Kernel 1: fused transpose + QKV projection.
//   x (B,C,S) f32  @  WqkvT (3C,C) bf16  ->  Q,K,V bf16 laid out (B,H,S,96)
// ---------------------------------------------------------------------------
__global__ __launch_bounds__(128) void qkv_proj_kernel(
    const float* __restrict__ x,
    const __bf16* __restrict__ WqkvT,   // (NC3, C) = W^T, bf16
    __bf16* __restrict__ qbuf,
    __bf16* __restrict__ kbuf,
    __bf16* __restrict__ vbuf)
{
    __shared__ __bf16 lds_a[16 * 32];   // [m][kk]

    const int b    = blockIdx.z;
    const int s0   = blockIdx.y * 16;
    const int n0b  = blockIdx.x * 256;
    const int tid  = threadIdx.x;
    const int wid  = tid >> 5;
    const int lane = tid & 31;
    const int m    = lane & 15;
    const int h    = lane >> 4;
    const int nw0  = n0b + wid * 64;

    v8f acc[4] = {};

    for (int k0 = 0; k0 < C_; k0 += 32) {
        // Cooperative staged transpose: 512 f32 -> bf16 into LDS.
        int idx = tid;
        #pragma unroll
        for (int i = 0; i < 4; ++i) {
            const int kk = idx >> 4;
            const int mm = idx & 15;
            const float v = x[(size_t)b * C_ * S_ + (size_t)(k0 + kk) * S_ + (s0 + mm)];
            lds_a[mm * 32 + kk] = (__bf16)v;
            idx += 128;
        }
        __syncthreads();

        v16bf a;   // A: element e -> K = 16h+e (contiguous)
        {
            const v8bf lo = *(const v8bf*)&lds_a[m * 32 + 16 * h];
            const v8bf hi = *(const v8bf*)&lds_a[m * 32 + 16 * h + 8];
            #pragma unroll
            for (int e = 0; e < 8; ++e) { a[e] = lo[e]; a[8 + e] = hi[e]; }
        }

        #pragma unroll
        for (int nt = 0; nt < 4; ++nt) {
            const int n0 = nw0 + nt * 16;
            // B fragment: lane col n=m, element e -> K = k0+16h+e: contiguous.
            const v16bf bm = *(const v16bf*)&WqkvT[(size_t)(n0 + m) * C_ + k0 + 16 * h];
            acc[nt] = wmma_acc(a, bm, acc[nt]);
        }
        __syncthreads();
    }

    #pragma unroll
    for (int nt = 0; nt < 4; ++nt) {
        const int n    = nw0 + nt * 16 + m;
        const int sel  = n / C_;
        const int rem  = n % C_;
        const int head = rem / HD_;
        const int d    = rem % HD_;
        __bf16* dst = (sel == 0) ? qbuf : (sel == 1) ? kbuf : vbuf;
        #pragma unroll
        for (int j = 0; j < 8; ++j) {
            const int srow = s0 + j + 8 * h;
            dst[((size_t)(b * NH_ + head) * S_ + srow) * HD_ + d] = (__bf16)acc[nt][j];
        }
    }
}

// ---------------------------------------------------------------------------
// Kernel 2: flash attention. 4 waves/block share one (b,h); K chunk staged by
// TDM (tensor_load_to_lds), V chunk staged by async global->LDS B128 copies
// then transposed in LDS so P@V B-fragments are contiguous ds_load vectors.
// ---------------------------------------------------------------------------
__global__ __launch_bounds__(128) void attention_kernel(
    const __bf16* __restrict__ qbuf,
    const __bf16* __restrict__ kbuf,
    const __bf16* __restrict__ vbuf,
    __bf16* __restrict__ attn)   // (B,S,C) bf16
{
    __shared__ __bf16 ldsk[32 * HD_];    // K chunk  [key][d]   6 KB
    __shared__ __bf16 ldsvr[32 * HD_];   // V raw    [key][d]   6 KB
    __shared__ __bf16 ldsvt[HD_ * 32];   // V transp [d][key]   6 KB
    __shared__ __bf16 ldsp[4][16 * 32];  // P per wave          4 KB

    const int tid  = threadIdx.x;
    const int wid  = tid >> 5;
    const int lane = tid & 31;
    const int m    = lane & 15;
    const int h    = lane >> 4;

    const int qt   = blockIdx.x * 4 + wid;  // 4 q-tiles per block, same (b,h)
    const int bh   = qt >> 7;
    const int q0   = (qt & 127) << 4;
    const int b    = bh >> 3;
    const int head = bh & 7;

    const size_t base = (size_t)bh * S_ * HD_;

    // Q A-fragments for the 3 d-chunks, kept in registers.
    v16bf aq[3];
    #pragma unroll
    for (int c = 0; c < 3; ++c) {
        const int dk = c * 32;
        const size_t rb = base + (size_t)(q0 + m) * HD_;
        const v8bf lo = *(const v8bf*)&qbuf[rb + dk + 16 * h];
        const v8bf hi = *(const v8bf*)&qbuf[rb + dk + 16 * h + 8];
        #pragma unroll
        for (int e = 0; e < 8; ++e) { aq[c][e] = lo[e]; aq[c][8 + e] = hi[e]; }
    }

    v8f   acc[6] = {};
    float mrow[8], lrow[8];
    #pragma unroll
    for (int j = 0; j < 8; ++j) { mrow[j] = -3.0e38f; lrow[j] = 0.0f; }

    for (int kc = 0; kc < S_; kc += 32) {
        if (kc + 32 < S_)
            __builtin_prefetch(&kbuf[base + (size_t)(kc + 32 + m) * HD_], 0, 1);

        // ---- stage K chunk (32x96 bf16, contiguous 6144B) via TDM ----
#if __has_builtin(__builtin_amdgcn_tensor_load_to_lds)
        if (wid == 0) {
            const uint64_t ga = (uint64_t)(uintptr_t)&kbuf[base + (size_t)kc * HD_];
            u32x4 g0 = {};
            i32x8 g1 = {};
            i32x4 g2 = {}, g3 = {};
            i32x8 g4 = {};
            g0[0] = 1u;                                        // count=1
            g0[1] = (unsigned)(uintptr_t)&ldsk[0];             // lds_addr
            g0[2] = (unsigned)ga;                              // global_addr[31:0]
            g0[3] = (unsigned)((ga >> 32) & 0x1FFFFFFu) | (2u << 30); // addr hi | type=2
            g1[0] = 0x00010000;                                // data_size = 2B
            g1[1] = (int)(3072u << 16);                        // tensor_dim0 lo16
            g1[2] = 0;                                         // tensor_dim0 hi / dim1 lo
            g1[3] = (int)(3072u << 16);                        // tile_dim0 = 3072
            g1[4] = 0;
            g1[5] = 3072;                                      // tensor_dim0_stride
            g1[6] = 0;
            g1[7] = 0;
            __builtin_amdgcn_tensor_load_to_lds(g0, g1, g2, g3, g4, 0);
            __builtin_amdgcn_s_wait_tensorcnt(0);
        }
#else
        {   // fallback: cooperative contiguous copy
            #pragma unroll
            for (int i = 0; i < 3; ++i) {
                const int g = tid + i * 128;                   // 384 x 8 elems
                *(v8bf*)&ldsk[g * 8] = *(const v8bf*)&kbuf[base + (size_t)kc * HD_ + g * 8];
            }
        }
#endif

        // ---- stage V chunk raw via async global->LDS B128 (ASYNCcnt) ----
        {
            const unsigned lbase = (unsigned)(uintptr_t)&ldsvr[0];
            #pragma unroll
            for (int i = 0; i < 3; ++i) {
                const int g = tid + i * 128;                   // 384 x 16B
                const unsigned    loff = lbase + (unsigned)(g * 16);
                const __bf16*     gp   = vbuf + base + (size_t)kc * HD_ + g * 8;
                asm volatile("global_load_async_to_lds_b128 %0, %1, off"
                             :: "v"(loff), "v"(gp) : "memory");
            }
            asm volatile("s_wait_asynccnt 0x0" ::: "memory");
        }
        __syncthreads();   // ldsk + ldsvr visible block-wide

        // ---- transpose V in LDS: [key][d] -> [d][key] ----
        #pragma unroll
        for (int i = 0; i < 24; ++i) {
            const int g   = tid + i * 128;   // 3072 elements
            const int d   = g >> 5;
            const int key = g & 31;
            ldsvt[g] = ldsvr[key * HD_ + d];
        }
        __syncthreads();   // ldsvt ready

        // ---- scores: S16x32 = Q @ K^T  (B fragments from LDS, contiguous) ----
        v8f sc0 = {}, sc1 = {};
        #pragma unroll
        for (int c = 0; c < 3; ++c) {
            const int dk = c * 32;
            const v16bf bk0 = *(const v16bf*)&ldsk[(m)      * HD_ + dk + 16 * h];
            const v16bf bk1 = *(const v16bf*)&ldsk[(16 + m) * HD_ + dk + 16 * h];
            sc0 = wmma_acc(aq[c], bk0, sc0);
            sc1 = wmma_acc(aq[c], bk1, sc1);
        }

        // ---- online softmax + stage P (per-wave LDS region) ----
        #pragma unroll
        for (int j = 0; j < 8; ++j) {
            const float rmax = rowmax16(fmaxf(sc0[j], sc1[j]));
            const float mnew = fmaxf(mrow[j], rmax);
            const float p0   = __expf(sc0[j] - mnew);
            const float p1   = __expf(sc1[j] - mnew);
            const float corr = __expf(mrow[j] - mnew);
            lrow[j] = lrow[j] * corr + rowsum16(p0 + p1);
            mrow[j] = mnew;
            #pragma unroll
            for (int t = 0; t < 6; ++t) acc[t][j] *= corr;
            ldsp[wid][(j + 8 * h) * 32 + m]      = (__bf16)p0;
            ldsp[wid][(j + 8 * h) * 32 + 16 + m] = (__bf16)p1;
        }
        // P written/read by the same wave only: DS ops are in-order per wave.

        v16bf ap;
        {
            const v8bf lo = *(const v8bf*)&ldsp[wid][m * 32 + 16 * h];
            const v8bf hi = *(const v8bf*)&ldsp[wid][m * 32 + 16 * h + 8];
            #pragma unroll
            for (int e = 0; e < 8; ++e) { ap[e] = lo[e]; ap[8 + e] = hi[e]; }
        }

        // ---- acc += P @ V: B fragments contiguous from transposed LDS ----
        #pragma unroll
        for (int t = 0; t < 6; ++t) {
            const v16bf bv = *(const v16bf*)&ldsvt[(t * 16 + m) * 32 + 16 * h];
            acc[t] = wmma_acc(ap, bv, acc[t]);
        }
        __syncthreads();   // before next iteration's staging overwrites
    }

    #pragma unroll
    for (int t = 0; t < 6; ++t) {
        const int col = head * HD_ + t * 16 + m;
        #pragma unroll
        for (int j = 0; j < 8; ++j) {
            const int srow = q0 + j + 8 * h;
            attn[((size_t)b * S_ + srow) * C_ + col] = (__bf16)(acc[t][j] / lrow[j]);
        }
    }
}

// ---------------------------------------------------------------------------
// Kernel 3: output projection with fused transpose:
//   attn (B,S,C) bf16 @ WoutT (C,C) bf16 -> out (B,C,S) f32
// ---------------------------------------------------------------------------
__global__ __launch_bounds__(128) void out_proj_kernel(
    const __bf16* __restrict__ attn,
    const __bf16* __restrict__ WoutT,   // (C, C) = W^T, bf16
    float* __restrict__ out)
{
    const int b    = blockIdx.z;
    const int s0   = blockIdx.y * 16;
    const int n0b  = blockIdx.x * 256;
    const int tid  = threadIdx.x;
    const int wid  = tid >> 5;
    const int lane = tid & 31;
    const int m    = lane & 15;
    const int h    = lane >> 4;
    const int nw0  = n0b + wid * 64;

    v8f acc[4] = {};

    for (int k0 = 0; k0 < C_; k0 += 32) {
        v16bf a;
        {
            const size_t rb = ((size_t)b * S_ + (s0 + m)) * C_;
            const v8bf lo = *(const v8bf*)&attn[rb + k0 + 16 * h];
            const v8bf hi = *(const v8bf*)&attn[rb + k0 + 16 * h + 8];
            #pragma unroll
            for (int e = 0; e < 8; ++e) { a[e] = lo[e]; a[8 + e] = hi[e]; }
        }
        #pragma unroll
        for (int nt = 0; nt < 4; ++nt) {
            const int n0 = nw0 + nt * 16;
            const v16bf bm = *(const v16bf*)&WoutT[(size_t)(n0 + m) * C_ + k0 + 16 * h];
            acc[nt] = wmma_acc(a, bm, acc[nt]);
        }
    }

    #pragma unroll
    for (int nt = 0; nt < 4; ++nt) {
        const int n = nw0 + nt * 16 + m;
        #pragma unroll
        for (int j = 0; j < 8; ++j) {
            out[(size_t)b * C_ * S_ + (size_t)n * S_ + (s0 + j + 8 * h)] = acc[nt][j];
        }
    }
}

// ---------------------------------------------------------------------------
extern "C" void kernel_launch(void* const* d_in, const int* in_sizes, int n_in,
                              void* d_out, int out_size, void* d_ws, size_t ws_size,
                              hipStream_t stream) {
    (void)in_sizes; (void)n_in; (void)out_size; (void)ws_size;

    const float* x    = (const float*)d_in[0];   // (2,768,2048)
    const float* Wqkv = (const float*)d_in[1];   // (768,2304)
    const float* Wout = (const float*)d_in[2];   // (768,768)
    float*       out  = (float*)d_out;           // (2,768,2048)

    const size_t qkv_elems  = (size_t)B_ * NH_ * S_ * HD_;   // 3,145,728
    const size_t attn_elems = (size_t)B_ * S_ * C_;          // 3,145,728

    __bf16* qbuf   = (__bf16*)d_ws;
    __bf16* kbuf   = qbuf + qkv_elems;
    __bf16* vbuf   = kbuf + qkv_elems;
    __bf16* attn   = vbuf + qkv_elems;
    __bf16* WqkvT  = attn + attn_elems;                      // (2304, 768)
    __bf16* WoutT  = WqkvT + (size_t)NC3_ * C_;              // (768, 768)

    wcvt_kernel<<<dim3((C_ * NC3_ + 255) / 256), 256, 0, stream>>>(Wqkv, WqkvT, C_, NC3_);
    wcvt_kernel<<<dim3((C_ * C_ + 255) / 256), 256, 0, stream>>>(Wout, WoutT, C_, C_);

    qkv_proj_kernel<<<dim3(NC3_ / 256, S_ / 16, B_), 128, 0, stream>>>(
        x, WqkvT, qbuf, kbuf, vbuf);

    attention_kernel<<<dim3((B_ * NH_ * (S_ / 16)) / 4), 128, 0, stream>>>(
        qbuf, kbuf, vbuf, attn);

    out_proj_kernel<<<dim3(C_ / 256, S_ / 16, B_), 128, 0, stream>>>(
        attn, WoutT, out);
}
